// AttentionSinkPrefill_4440996184664
// MI455X (gfx1250) — matrix operand order repaired and checked by reading, not compiled, not run on hardware
//
#include <hip/hip_runtime.h>
#include <hip/hip_bf16.h>

typedef __attribute__((ext_vector_type(16))) __bf16 v16bf;
typedef __attribute__((ext_vector_type(8)))  float  v8f;

union Frag16 {
  v16bf v;
  unsigned int u32[8];
  unsigned short u16[16];
};

__device__ __forceinline__ unsigned short f2bf(float f) {
  unsigned int u = __float_as_uint(f);
  u += 0x7FFFu + ((u >> 16) & 1u);   // round-to-nearest-even
  return (unsigned short)(u >> 16);
}

// Async global->LDS copy of 16 bytes per lane (CDNA5, ASYNCcnt-tracked).
__device__ __forceinline__ void async_b128(unsigned lds_off, const void* gaddr) {
  asm volatile("global_load_async_to_lds_b128 %0, %1, off"
               :: "v"(lds_off), "v"(gaddr) : "memory");
}
__device__ __forceinline__ void wait_async0() {
  asm volatile("s_wait_asynccnt 0" ::: "memory");
}

// ---------------------------------------------------------------------------
// Elementwise fp32 -> bf16 (one pass; n divisible by 4).
// ---------------------------------------------------------------------------
__global__ __launch_bounds__(256)
void cvt_bf16(const float* __restrict__ in, unsigned short* __restrict__ out) {
  const int i = blockIdx.x * 256 + threadIdx.x;
  float4 f = ((const float4*)in)[i];
  uint2 r;
  r.x = (unsigned)f2bf(f.x) | ((unsigned)f2bf(f.y) << 16);
  r.y = (unsigned)f2bf(f.z) | ((unsigned)f2bf(f.w) << 16);
  ((uint2*)out)[i] = r;
}

// ---------------------------------------------------------------------------
// fp32 [K][N] -> bf16 [N][K] (transpose-convert for weight matrices).
// ---------------------------------------------------------------------------
__global__ __launch_bounds__(256)
void cvt_bf16_t(const float* __restrict__ in, unsigned short* __restrict__ out,
                int K, int N) {
  const int idx = blockIdx.x * 256 + threadIdx.x;
  const int k = idx / N, n = idx % N;
  out[(size_t)n * K + k] = f2bf(in[idx]);
}

// ---------------------------------------------------------------------------
// C(MxN) = scale * A(MxK) @ B(KxN), with A bf16 row-major [M][K] and B given
// PRE-TRANSPOSED bf16 [N][K].  128 threads = 4 waves, block tile 64x64,
// K step 32.  Tiles staged via global_load_async_to_lds_b128.
// OUT_T stores C transposed ([N][M]) — used for V so attention reads [hd][tok].
// ---------------------------------------------------------------------------
template <bool OUT_F32, bool OUT_T>
__global__ __launch_bounds__(128)
void wmma_gemm(const unsigned short* __restrict__ A,
               const unsigned short* __restrict__ Bt,
               void* __restrict__ Cp, int M, int N, int K, float scale) {
  __shared__ unsigned short ldsA[64][40];  // [m][k], 80B row stride (16B-aligned)
  __shared__ unsigned short ldsB[64][40];  // [n][k]
  const int tid  = threadIdx.x;
  const int wave = tid >> 5;
  const int lane = tid & 31;
  const int l16  = lane & 15;
  const int hi   = lane >> 4;
  const int bm   = blockIdx.y * 64;
  const int bn   = blockIdx.x * 64;

  const int r = tid >> 1;            // staging row 0..63
  const int c = (tid & 1) * 16;      // 16-element (32B) half-row

  v8f acc[4];
  #pragma unroll
  for (int j = 0; j < 4; ++j)
    #pragma unroll
    for (int e = 0; e < 8; ++e) acc[j][e] = 0.0f;

  for (int k0 = 0; k0 < K; k0 += 32) {
    // ---- async-stage A tile (64x32) and Bt tile (64x32) ----
    {
      const unsigned short* ga = A + (size_t)(bm + r) * K + k0 + c;
      const unsigned la = (unsigned)(size_t)&ldsA[r][c];
      async_b128(la,      ga);
      async_b128(la + 16, ga + 8);
      const unsigned short* gb = Bt + (size_t)(bn + r) * K + k0 + c;
      const unsigned lb = (unsigned)(size_t)&ldsB[r][c];
      async_b128(lb,      gb);
      async_b128(lb + 16, gb + 8);
    }
    wait_async0();
    __syncthreads();

    // A fragment: 16-bit A 16x32 layout — lane row = lane&15,
    // k = (v>=4?16:0) + 8*hi + 2*(v&3) + {0,1}
    Frag16 a;
    const int arow = wave * 16 + l16;
    #pragma unroll
    for (int v = 0; v < 8; ++v) {
      const int kk = ((v & 4) ? 16 : 0) + hi * 8 + (v & 3) * 2;
      a.u32[v] = *(const unsigned int*)&ldsA[arow][kk];
    }
    // B fragments: 16-bit B 32x16 layout — lane col = lane&15, k = (v>=4?16:0) + 8*hi + 2*(v&3)
    #pragma unroll
    for (int j = 0; j < 4; ++j) {
      Frag16 bfr;
      const int col = j * 16 + l16;
      #pragma unroll
      for (int v = 0; v < 8; ++v)
        bfr.u32[v] = *(const unsigned int*)&ldsB[col][((v & 4) ? 16 : 0) + hi * 8 + (v & 3) * 2];
      acc[j] = __builtin_amdgcn_wmma_f32_16x16x32_bf16(
          false, a.v, false, bfr.v, (short)0, acc[j], false, false);
    }
    __syncthreads();
  }

  // Epilogue: f32 C/D layout — elem e is row e + 8*hi, col = lane&15
  #pragma unroll
  for (int j = 0; j < 4; ++j) {
    #pragma unroll
    for (int e = 0; e < 8; ++e) {
      const int row = bm + wave * 16 + e + hi * 8;
      const int col = bn + j * 16 + l16;
      const float val = acc[j][e] * scale;
      const size_t idx = OUT_T ? ((size_t)col * M + row) : ((size_t)row * N + col);
      if (OUT_F32) ((float*)Cp)[idx] = val;
      else         ((unsigned short*)Cp)[idx] = f2bf(val);
    }
  }
}

// ---------------------------------------------------------------------------
// Flash-style sink+sliding-window attention. One wave per 16-query tile.
// Q pre-scaled by 1/sqrt(64). V is pre-transposed: Vt[kv*64+hd][b*T+tok].
// ---------------------------------------------------------------------------
__global__ __launch_bounds__(128)
void wmma_attn(const unsigned short* __restrict__ Qbf,
               const unsigned short* __restrict__ Kbf,
               const unsigned short* __restrict__ Vt,
               unsigned short* __restrict__ Ybf,
               int Bb, int T) {
  constexpr int H = 16, KVH = 4, HD = 64, SINK = 4, WIN = 256;
  const int Mtok = Bb * T;
  __shared__ unsigned short pbuf[4][16][36];   // per-wave P relayout buffer
  const int tid = threadIdx.x, wave = tid >> 5, lane = tid & 31;
  const int l16 = lane & 15, hi = lane >> 4;
  const int gw = blockIdx.x * 4 + wave;
  const int qtiles = T / 16;
  const int qt = gw % qtiles;
  const int h  = (gw / qtiles) % H;
  const int b  = gw / (qtiles * H);
  if (b >= Bb) return;
  const int t0  = qt * 16;
  const int kvh = h / (H / KVH);

  // Q fragments (A-layout): per lane two contiguous 16B runs per 32-k chunk
  Frag16 qa[2];
  const unsigned short* Qrow = Qbf + (size_t)(b * T + t0 + l16) * (H * HD) + h * HD;
  #pragma unroll
  for (int cc = 0; cc < 2; ++cc) {
    uint4 q0 = *(const uint4*)(Qrow + cc * 32 + hi * 8);
    uint4 q1 = *(const uint4*)(Qrow + cc * 32 + 16 + hi * 8);
    qa[cc].u32[0] = q0.x; qa[cc].u32[1] = q0.y; qa[cc].u32[2] = q0.z; qa[cc].u32[3] = q0.w;
    qa[cc].u32[4] = q1.x; qa[cc].u32[5] = q1.y; qa[cc].u32[6] = q1.z; qa[cc].u32[7] = q1.w;
  }

  v8f o[4];
  #pragma unroll
  for (int j = 0; j < 4; ++j)
    #pragma unroll
    for (int e = 0; e < 8; ++e) o[j][e] = 0.0f;
  float m[8], l[8];
  #pragma unroll
  for (int e = 0; e < 8; ++e) { m[e] = -__builtin_inff(); l[e] = 0.0f; }

  const int kbend = (t0 + 15) >> 5;
  int lo = t0 - (WIN - 1); if (lo < 0) lo = 0;
  const int kbw0 = lo >> 5;

  for (int kb = 0; kb <= kbend; ++kb) {
    if (kb != 0 && kb < kbw0) continue;   // skip blocks outside sink+window
    const int j0 = kb * 32;

    // S = Q @ K^T for 32 keys: B-frag lane col = key, k = head-dim chunk
    v8f s[2];
    #pragma unroll
    for (int n = 0; n < 2; ++n) {
      const unsigned short* Krow =
          Kbf + (size_t)(b * T + j0 + n * 16 + l16) * (KVH * HD) + kvh * HD;
      Frag16 kf[2];
      #pragma unroll
      for (int cc = 0; cc < 2; ++cc) {
        uint4 k0 = *(const uint4*)(Krow + cc * 32 + hi * 8);
        uint4 k1 = *(const uint4*)(Krow + cc * 32 + 16 + hi * 8);
        kf[cc].u32[0] = k0.x; kf[cc].u32[1] = k0.y; kf[cc].u32[2] = k0.z; kf[cc].u32[3] = k0.w;
        kf[cc].u32[4] = k1.x; kf[cc].u32[5] = k1.y; kf[cc].u32[6] = k1.z; kf[cc].u32[7] = k1.w;
      }
      v8f sv;
      #pragma unroll
      for (int e = 0; e < 8; ++e) sv[e] = 0.0f;
      sv = __builtin_amdgcn_wmma_f32_16x16x32_bf16(false, qa[0].v, false, kf[0].v, (short)0, sv, false, false);
      sv = __builtin_amdgcn_wmma_f32_16x16x32_bf16(false, qa[1].v, false, kf[1].v, (short)0, sv, false, false);
      s[n] = sv;
    }

    // Mask + online softmax (16-lane row reductions stay in half-wave)
    #pragma unroll
    for (int e = 0; e < 8; ++e) {
      const int i  = t0 + e + hi * 8;
      const int ja = j0 + l16;
      const int jb = j0 + 16 + l16;
      float v0 = ((ja <= i) && ((ja < SINK) || (i - ja < WIN))) ? s[0][e] : -__builtin_inff();
      float v1 = ((jb <= i) && ((jb < SINK) || (i - jb < WIN))) ? s[1][e] : -__builtin_inff();
      float rm = fmaxf(v0, v1);
      rm = fmaxf(rm, __shfl_xor(rm, 1, 32));
      rm = fmaxf(rm, __shfl_xor(rm, 2, 32));
      rm = fmaxf(rm, __shfl_xor(rm, 4, 32));
      rm = fmaxf(rm, __shfl_xor(rm, 8, 32));
      const float mnew = fmaxf(m[e], rm);
      const float al = __expf(m[e] - mnew);   // first block: exp(-inf)=0
      const float p0 = __expf(v0 - mnew);
      const float p1 = __expf(v1 - mnew);
      float rs = p0 + p1;
      rs += __shfl_xor(rs, 1, 32);
      rs += __shfl_xor(rs, 2, 32);
      rs += __shfl_xor(rs, 4, 32);
      rs += __shfl_xor(rs, 8, 32);
      l[e] = l[e] * al + rs;
      m[e] = mnew;
      const int rr = e + hi * 8;
      pbuf[wave][rr][l16]      = f2bf(p0);
      pbuf[wave][rr][16 + l16] = f2bf(p1);
      #pragma unroll
      for (int j = 0; j < 4; ++j) o[j][e] *= al;
    }

    asm volatile("s_wait_dscnt 0" ::: "memory");  // P stores -> A-frag reads

    // P as A fragment (16x32 bf16)
    Frag16 pa;
    #pragma unroll
    for (int v = 0; v < 8; ++v) {
      const int kk = hi * 16 + v * 2;
      pa.u32[v] = *(const unsigned int*)&pbuf[wave][l16][kk];
    }

    // O += P @ V : B-frag lane col = head-dim; keys contiguous in Vt rows
    #pragma unroll
    for (int nc = 0; nc < 4; ++nc) {
      const unsigned short* Vcol =
          Vt + (size_t)(kvh * HD + nc * 16 + l16) * Mtok + b * T + j0 + hi * 16;
      uint4 w0 = ((const uint4*)Vcol)[0];
      uint4 w1 = ((const uint4*)Vcol)[1];
      Frag16 vb;
      vb.u32[0] = w0.x; vb.u32[1] = w0.y; vb.u32[2] = w0.z; vb.u32[3] = w0.w;
      vb.u32[4] = w1.x; vb.u32[5] = w1.y; vb.u32[6] = w1.z; vb.u32[7] = w1.w;
      o[nc] = __builtin_amdgcn_wmma_f32_16x16x32_bf16(false, pa.v, false, vb.v, (short)0, o[nc], false, false);
    }
  }

  // Normalize and store Y (bf16, [B,T,H*64])
  #pragma unroll
  for (int e = 0; e < 8; ++e) {
    const float inv = 1.0f / l[e];
    const int rr = e + hi * 8;
    unsigned short* Yr = Ybf + (size_t)(b * T + t0 + rr) * (H * HD) + h * HD + l16;
    #pragma unroll
    for (int nc = 0; nc < 4; ++nc)
      Yr[nc * 16] = f2bf(o[nc][e] * inv);
  }
}

// ---------------------------------------------------------------------------
extern "C" void kernel_launch(void* const* d_in, const int* in_sizes, int n_in,
                              void* d_out, int out_size, void* d_ws, size_t ws_size,
                              hipStream_t stream) {
  (void)n_in; (void)out_size; (void)ws_size;
  const float* X  = (const float*)d_in[0];
  const float* Wq = (const float*)d_in[1];
  const float* Wk = (const float*)d_in[2];
  const float* Wv = (const float*)d_in[3];
  const float* Wo = (const float*)d_in[4];

  const int D = 1024, H = 16, KVH = 4, HD = 64, T = 2048;
  const int M  = in_sizes[0] / D;   // B*T
  const int Bb = M / T;

  unsigned char* ws = (unsigned char*)d_ws;
  size_t off = 0;
  auto alloc = [&](size_t elems) {
    unsigned short* p = (unsigned short*)(ws + off);
    off += elems * 2;
    off = (off + 255) & ~(size_t)255;
    return p;
  };
  unsigned short* Xbf = alloc((size_t)M * D);
  unsigned short* Wqt = alloc((size_t)D * (H * HD));
  unsigned short* Wkt = alloc((size_t)D * (KVH * HD));
  unsigned short* Wvt = alloc((size_t)D * (KVH * HD));
  unsigned short* Wot = alloc((size_t)(H * HD) * D);
  unsigned short* Qbf = alloc((size_t)M * (H * HD));
  unsigned short* Kbf = alloc((size_t)M * (KVH * HD));
  unsigned short* Vtb = alloc((size_t)(KVH * HD) * M);   // transposed V
  unsigned short* Ybf = alloc((size_t)M * (H * HD));

  dim3 blk(128);
  // One-pass conversions / weight transposes (bandwidth-trivial: ~42 MB)
  cvt_bf16<<<dim3((M * D) / 1024), dim3(256), 0, stream>>>(X, Xbf);
  cvt_bf16_t<<<dim3((D * H * HD) / 256), dim3(256), 0, stream>>>(Wq, Wqt, D, H * HD);
  cvt_bf16_t<<<dim3((D * KVH * HD) / 256), dim3(256), 0, stream>>>(Wk, Wkt, D, KVH * HD);
  cvt_bf16_t<<<dim3((D * KVH * HD) / 256), dim3(256), 0, stream>>>(Wv, Wvt, D, KVH * HD);
  cvt_bf16_t<<<dim3((H * HD * D) / 256), dim3(256), 0, stream>>>(Wo, Wot, H * HD, D);

  // Q = (X @ Wq) / 8 ; K = X @ Wk ; V^T = (X @ Wv)^T
  wmma_gemm<false, false><<<dim3((H * HD) / 64, M / 64), blk, 0, stream>>>(Xbf, Wqt, Qbf, M, H * HD, D, 0.125f);
  wmma_gemm<false, false><<<dim3((KVH * HD) / 64, M / 64), blk, 0, stream>>>(Xbf, Wkt, Kbf, M, KVH * HD, D, 1.0f);
  wmma_gemm<false, true ><<<dim3((KVH * HD) / 64, M / 64), blk, 0, stream>>>(Xbf, Wvt, Vtb, M, KVH * HD, D, 1.0f);

  const int nwaves = Bb * H * (T / 16);
  wmma_attn<<<dim3(nwaves / 4), blk, 0, stream>>>(Qbf, Kbf, Vtb, Ybf, Bb, T);

  // out = Y @ Wo  (f32 output)
  wmma_gemm<true, false><<<dim3(D / 64, M / 64), blk, 0, stream>>>(Ybf, Wot, (float*)d_out, M, D, H * HD, 1.0f);
}
